// SMB_13718125543465
// MI455X (gfx1250) — compile-verified, matrix operand order checked by reading, not run on hardware
//
#include <hip/hip_runtime.h>

// ---------- types ----------
typedef __bf16       v16bf __attribute__((ext_vector_type(16)));
typedef float        v8f   __attribute__((ext_vector_type(8)));
typedef unsigned int v4u   __attribute__((ext_vector_type(4)));

union Frag {              // one WMMA 16-bit operand: 16 bf16 = 8 VGPRs = 2x b128
    v16bf v;
    v4u   q[2];
};

// Problem constants
#define Bn   16
#define Cn   64
#define Hn   128
#define Wn   128
#define HWn  (Hn*Wn)            // 16384
#define Kc   576                // 9 taps * 64 ch
#define OUTE (Bn*Cn*HWn)        // 16777216 fp32 elements of primary output

// ---------------------------------------------------------------------------
// 1) gumbel softmax gates (also second tuple output)
// ---------------------------------------------------------------------------
__global__ __launch_bounds__(256) void prep_kernel(const float* __restrict__ gum,
                                                   const float* __restrict__ par,
                                                   float* __restrict__ chm,
                                                   float* __restrict__ out_tail)
{
    int t = threadIdx.x;                       // t = c*4 + i  (256 pairs)
    float u0 = gum[t*2+0], u1 = gum[t*2+1];
    float g0 = -logf(-logf(u0));
    float g1 = -logf(-logf(u1));
    float l0 = par[t*2+0] + g0;
    float l1 = par[t*2+1] + g1;
    float mx = fmaxf(l0, l1);
    float e0 = expf(l0 - mx), e1 = expf(l1 - mx);
    float inv = 1.0f / (e0 + e1);
    float m0 = e0 * inv, m1 = e1 * inv;
    chm[t*2+0] = m0;  chm[t*2+1] = m1;
    out_tail[t*2+0] = m0;  out_tail[t*2+1] = m1;
}

// ---------------------------------------------------------------------------
// 2) fold channel gates into bf16 weight banks, layout [o][tap*64+c]
//    (A-fragment friendly: per lane two contiguous 8-elem runs) + pack wc
// ---------------------------------------------------------------------------
__global__ __launch_bounds__(256) void pack_kernel(const float* __restrict__ w0,
                                                   const float* __restrict__ w1,
                                                   const float* __restrict__ w2,
                                                   const float* __restrict__ w3,
                                                   const float* __restrict__ wc,
                                                   const float* __restrict__ chm,
                                                   __bf16* __restrict__ wd_all,
                                                   __bf16* __restrict__ ws_all,
                                                   __bf16* __restrict__ wcp_all)
{
    int T = blockIdx.x * 256 + threadIdx.x;
    if (T < 4*Cn*Kc) {                              // 147456 conv weights
        int i   = T / (Cn*Kc);
        int rem = T % (Cn*Kc);
        int o   = rem / Kc;
        int kk  = rem % Kc;
        int tap = kk / 64;
        int c   = kk % 64;
        const float* w = (i==0) ? w0 : (i==1) ? w1 : (i==2) ? w2 : w3;
        float wv = w[(o*64 + c)*9 + tap];
        float gd = (i==0) ? 1.0f : chm[c*8 + i*2 + 0];
        float gs = (i==0) ? 0.0f : chm[c*8 + i*2 + 1];
        wd_all[(size_t)i*(Cn*Kc) + o*Kc + kk] = (__bf16)(wv * gd);
        ws_all[(size_t)i*(Cn*Kc) + o*Kc + kk] = (__bf16)(wv * gs);
    } else {                                        // 16384 wc weights
        int T2 = T - 4*Cn*Kc;
        int i  = T2 / 4096;
        int r  = T2 % 4096;
        int o  = r / 64;
        int c  = r % 64;
        wcp_all[i*4096 + o*64 + c] = (__bf16)wc[o*256 + i*64 + c];
    }
}

// ---------------------------------------------------------------------------
// 3) x0 NCHW f32  ->  NHWC bf16
// ---------------------------------------------------------------------------
__global__ __launch_bounds__(256) void cvt_kernel(const float* __restrict__ x0,
                                                  __bf16* __restrict__ dst)
{
    size_t T = (size_t)blockIdx.x * 256 + threadIdx.x;   // < 16777216
    int p = (int)(T & (HWn-1));
    int c = (int)((T >> 14) & 63);
    int b = (int)(T >> 20);
    dst[((size_t)b*HWn + p)*64 + c] = (__bf16)x0[T];
}

// ---------------------------------------------------------------------------
// 4) out = bias (fp32 accumulator for the fused 1x1 conv)
// ---------------------------------------------------------------------------
__global__ __launch_bounds__(256) void init_kernel(const float* __restrict__ bc,
                                                   float* __restrict__ out)
{
    size_t T = (size_t)blockIdx.x * 256 + threadIdx.x;
    int o = (int)((T >> 14) & 63);
    out[T] = bc[o];
}

// ---------------------------------------------------------------------------
// 5) implicit-GEMM 3x3 conv, dual path (dense+sparse gated weights),
//    fused gate/spa/relu epilogue.  One block = one (batch, 2-row strip).
//    LDS tile: 4 rows x 130 cols x 64 ch bf16 (zero-padded halo columns).
//    K (taps) is the OUTER loop: each weight A-fragment is loaded once and
//    reused across all 8 n-tiles whose accumulators stay resident (128 VGPRs),
//    keeping the wave under the 256-VGPR direct window.
// ---------------------------------------------------------------------------
__global__ __launch_bounds__(256) void conv_kernel(const __bf16* __restrict__ fin,
                                                   __bf16* __restrict__ fout,
                                                   const __bf16* __restrict__ wd,
                                                   const __bf16* __restrict__ wsp,
                                                   const float* __restrict__ spa,
                                                   const float* __restrict__ chm,
                                                   int layer)
{
    extern __shared__ __bf16 tile[];                 // [4][130][64]
    const int b  = blockIdx.x >> 6;                  // batch
    const int rb = blockIdx.x & 63;                  // 2-row strip
    const int y0 = rb * 2;
    const int t  = threadIdx.x;

    // ---- stage input rows y0-1..y0+2 into LDS (4096 x b128 chunks) ----
    for (int j = 0; j < 16; ++j) {
        int q   = t + j*256;                         // 0..4095
        int ch  = (q & 7) * 8;
        int col = (q >> 3) & 127;
        int row = q >> 10;                           // 0..3
        int gy  = y0 - 1 + row;
        v4u val = (v4u)0u;
        if (gy >= 0 && gy < Hn)
            val = *(const v4u*)(fin + (((size_t)b*HWn + gy*Wn + col)*64 + ch));
        *(v4u*)(tile + ((row*130 + col + 1)*64 + ch)) = val;
    }
    if (t < 64) {                                    // zero halo columns 0 and 129
        int row = t >> 4, colp = (t >> 3) & 1, ch = (t & 7) * 8;
        *(v4u*)(tile + ((row*130 + (colp ? 129 : 0))*64 + ch)) = (v4u)0u;
    }
    __syncthreads();

    const int wave  = t >> 5, lane = t & 31;
    const int mtile = wave & 3;                      // output-channel tile
    const int half  = lane >> 4, ln = lane & 15;
    const int m     = mtile*16 + ln;                 // A-fragment row

    // per-lane spatial coordinates of the 8 n-tiles this wave owns
    const int nt0 = (wave >> 2) * 8;
    int lbase0[8];                                   // (r_out*130 + x)*64
    #pragma unroll
    for (int nt = 0; nt < 8; ++nt) {
        int idx = (nt0 + nt)*16 + ln;                // 0..255 within strip
        lbase0[nt] = ((idx >> 7)*130 + (idx & 127)) * 64;
    }

    v8f accd[8], accs[8];
    #pragma unroll
    for (int nt = 0; nt < 8; ++nt) { accd[nt] = (v8f)0.0f; accs[nt] = (v8f)0.0f; }

    const __bf16* wdl = wd  + m*Kc + half*8;
    const __bf16* wsl = wsp + m*Kc + half*8;

    #pragma unroll 1
    for (int tap = 0; tap < 9; ++tap) {
        const int dy = tap / 3, dx = tap % 3;
        const int toff = (dy*130 + dx) * 64;
        #pragma unroll
        for (int cb = 0; cb < 2; ++cb) {
            Frag ad, as;                             // weight A-fragments (shared by 8 tiles)
            const int kb = tap*64 + cb*32;
            ad.q[0] = *(const v4u*)(wdl + kb);
            ad.q[1] = *(const v4u*)(wdl + kb + 16);
            as.q[0] = *(const v4u*)(wsl + kb);
            as.q[1] = *(const v4u*)(wsl + kb + 16);
            #pragma unroll
            for (int nt = 0; nt < 8; ++nt) {
                Frag bf;
                const int lb = lbase0[nt] + toff + cb*32 + half*8;
                bf.q[0] = *(const v4u*)(tile + lb);
                bf.q[1] = *(const v4u*)(tile + lb + 16);
                accd[nt] = __builtin_amdgcn_wmma_f32_16x16x32_bf16(
                               false, ad.v, false, bf.v, (short)0, accd[nt], false, false);
                accs[nt] = __builtin_amdgcn_wmma_f32_16x16x32_bf16(
                               false, as.v, false, bf.v, (short)0, accs[nt], false, false);
            }
        }
    }

    // fused epilogue: fea = relu(out_d*(ms*spa+md) + out_s*spa)
    float gmd[8], gms[8];
    #pragma unroll
    for (int r = 0; r < 8; ++r) {
        int o  = mtile*16 + half*8 + r;
        gmd[r] = chm[o*8 + layer*2 + 0];
        gms[r] = chm[o*8 + layer*2 + 1];
    }
    #pragma unroll
    for (int nt = 0; nt < 8; ++nt) {
        int idx   = (nt0 + nt)*16 + ln;
        int p     = (y0 + (idx >> 7))*Wn + (idx & 127);
        float sp  = spa[(size_t)b*HWn + p];
        #pragma unroll
        for (int r = 0; r < 8; ++r) {
            int o = mtile*16 + half*8 + r;
            float v = accd[nt][r] * (gms[r]*sp + gmd[r]) + accs[nt][r]*sp;
            v = fmaxf(v, 0.0f);
            fout[((size_t)b*HWn + p)*64 + o] = (__bf16)v;
        }
    }
}

// ---------------------------------------------------------------------------
// 6) fused 1x1-conv update:  out[b,o,p] += sum_c wcp[o,c] * fea[b,p,c]
//    (K=64 -> two WMMAs; fp32 C loaded from d_out, chained, stored back)
// ---------------------------------------------------------------------------
__global__ __launch_bounds__(256) void accum_kernel(const __bf16* __restrict__ fea,
                                                    const __bf16* __restrict__ wcp,
                                                    float* __restrict__ out)
{
    int t = threadIdx.x, wave = t >> 5, lane = t & 31;
    int g     = blockIdx.x * 8 + wave;               // 65536 tiles
    int mtile = g & 3;
    int nt    = g >> 2;
    int b     = nt >> 10;
    int p     = (nt & 1023) * 16 + (lane & 15);
    int half  = lane >> 4, ln = lane & 15;

    v8f c;
    #pragma unroll
    for (int r = 0; r < 8; ++r) {
        int o = mtile*16 + half*8 + r;
        c[r] = out[((size_t)b*64 + o)*HWn + p];
    }
    const __bf16* bbase = fea + ((size_t)b*HWn + p)*64;
    #pragma unroll
    for (int cb = 0; cb < 2; ++cb) {
        Frag a, bf;
        bf.q[0] = *(const v4u*)(bbase + cb*32 + half*8);
        bf.q[1] = *(const v4u*)(bbase + cb*32 + 16 + half*8);
        int m = mtile*16 + ln;
        const __bf16* abase = wcp + m*64 + cb*32 + half*8;
        a.q[0] = *(const v4u*)(abase);
        a.q[1] = *(const v4u*)(abase + 16);
        c = __builtin_amdgcn_wmma_f32_16x16x32_bf16(
                false, a.v, false, bf.v, (short)0, c, false, false);
    }
    #pragma unroll
    for (int r = 0; r < 8; ++r) {
        int o = mtile*16 + half*8 + r;
        out[((size_t)b*64 + o)*HWn + p] = c[r];
    }
}

// ---------------------------------------------------------------------------
extern "C" void kernel_launch(void* const* d_in, const int* in_sizes, int n_in,
                              void* d_out, int out_size, void* d_ws, size_t ws_size,
                              hipStream_t stream) {
    const float* x0   = (const float*)d_in[0];
    const float* spa  = (const float*)d_in[1];
    const float* gum  = (const float*)d_in[2];
    const float* par  = (const float*)d_in[3];
    const float* w0   = (const float*)d_in[4];
    const float* w1   = (const float*)d_in[5];
    const float* w2   = (const float*)d_in[6];
    const float* w3   = (const float*)d_in[7];
    const float* wc   = (const float*)d_in[8];
    const float* bc   = (const float*)d_in[9];
    float* out        = (float*)d_out;              // [16,64,128,128] then ch_mask[512]
    float* out_tail   = out + OUTE;

    // workspace carve-up
    char* ws = (char*)d_ws;
    float*  chm     = (float*)(ws + 0);                         //   2 KB
    __bf16* wd_all  = (__bf16*)(ws + 4096);                     // 288 KB
    __bf16* ws_all  = (__bf16*)(ws + 4096 + 294912);            // 288 KB
    __bf16* wcp_all = (__bf16*)(ws + 4096 + 2*294912);          //  32 KB
    __bf16* feaA    = (__bf16*)(ws + 4096 + 2*294912 + 32768);  //  32 MB
    __bf16* feaB    = feaA + (size_t)Bn*HWn*Cn;                 //  32 MB

    const size_t lds_bytes = 4 * 130 * 64 * sizeof(__bf16);     // 66560

    prep_kernel<<<1, 256, 0, stream>>>(gum, par, chm, out_tail);
    pack_kernel<<<640, 256, 0, stream>>>(w0, w1, w2, w3, wc, chm,
                                         wd_all, ws_all, wcp_all);
    cvt_kernel<<<OUTE/256, 256, 0, stream>>>(x0, feaA);
    init_kernel<<<OUTE/256, 256, 0, stream>>>(bc, out);

    const int conv_grid  = Bn * (Hn/2);     // 1024
    const int accum_grid = 8192;            // 65536 tiles / 8 waves

    // layer 0: feaA -> feaB (ws bank is zero-gated for layer 0)
    conv_kernel<<<conv_grid, 256, lds_bytes, stream>>>(
        feaA, feaB, wd_all + 0*(size_t)Cn*Kc, ws_all + 0*(size_t)Cn*Kc, spa, chm, 0);
    accum_kernel<<<accum_grid, 256, 0, stream>>>(feaB, wcp_all + 0*4096, out);

    // layer 1: feaB -> feaA
    conv_kernel<<<conv_grid, 256, lds_bytes, stream>>>(
        feaB, feaA, wd_all + 1*(size_t)Cn*Kc, ws_all + 1*(size_t)Cn*Kc, spa, chm, 1);
    accum_kernel<<<accum_grid, 256, 0, stream>>>(feaA, wcp_all + 1*4096, out);

    // layer 2: feaA -> feaB
    conv_kernel<<<conv_grid, 256, lds_bytes, stream>>>(
        feaA, feaB, wd_all + 2*(size_t)Cn*Kc, ws_all + 2*(size_t)Cn*Kc, spa, chm, 2);
    accum_kernel<<<accum_grid, 256, 0, stream>>>(feaB, wcp_all + 2*4096, out);

    // layer 3: feaB -> feaA
    conv_kernel<<<conv_grid, 256, lds_bytes, stream>>>(
        feaB, feaA, wd_all + 3*(size_t)Cn*Kc, ws_all + 3*(size_t)Cn*Kc, spa, chm, 3);
    accum_kernel<<<accum_grid, 256, 0, stream>>>(feaA, wcp_all + 3*4096, out);
}
